// BasicGraphModel_64484638982496
// MI455X (gfx1250) — matrix operand order repaired and checked by reading, not compiled, not run on hardware
//
#include <hip/hip_runtime.h>
#include <hip/hip_bf16.h>

typedef __attribute__((ext_vector_type(16))) _Float16 v16h;
typedef __attribute__((ext_vector_type(8)))  float    v8f;

#define N_NODES 100000
#define N_EDGES 3200000
#define C_IN    932
#define C_HID   32
#define C_OUT   16

// ---------------- degree / normalization ----------------
__global__ void k_deg_init(float* deg) {
    int i = blockIdx.x * blockDim.x + threadIdx.x;
    if (i < N_NODES) deg[i] = 1.0f;   // self-loop contribution
}
__global__ void k_deg_edges(const int* __restrict__ dst, float* deg) {
    int e = blockIdx.x * blockDim.x + threadIdx.x;
    if (e < N_EDGES) atomicAdd(&deg[dst[e]], 1.0f);
}
__global__ void k_dinv(float* deg) {
    int i = blockIdx.x * blockDim.x + threadIdx.x;
    if (i < N_NODES) {
        float d = deg[i];
        deg[i] = (d > 0.0f) ? rsqrtf(d) : 0.0f;
    }
}

// ---------------- weight packing into WMMA B-fragment layout ----------------
// B fragment (16-bit, 32x16 per tile): lane L holds column N = L&15,
// K half = L>>4; element j of v16h is K = (L>>4)*8 + j (j<8), 16 + (L>>4)*8 + (j-8) (j>=8) within the chunk.
// Packed layout: pb[((chunk*NT + nt)*32 + lane)*16 + j]
__global__ void k_pack_B(const float* __restrict__ W, _Float16* pb,
                         int K, int C, int chunks) {
    int NT = C >> 4;
    int total = chunks * NT * 512;
    int idx = blockIdx.x * blockDim.x + threadIdx.x;
    if (idx >= total) return;
    int j     = idx & 15;
    int lane  = (idx >> 4) & 31;
    int nt    = (idx >> 9) % NT;
    int chunk = idx / (NT << 9);
    int n = nt * 16 + (lane & 15);
    int k = chunk * 32 + ((lane >> 4) << 3) + j + ((j >= 8) ? 8 : 0);
    pb[idx] = (_Float16)((k < K) ? W[(size_t)k * C + n] : 0.0f);
}

// ---------------- GEMM: H[M x (NT*16)] = A[M x K] * B, f16 WMMA, f32 acc ----
// Each wave owns MT consecutive 16-row M tiles so every B fragment load is
// reused MT times (cuts L2 B-traffic by MT). A is streamed fp32 from HBM and
// converted to f16 in-register per the 16-bit A fragment layout
// (lane row = lane&15, K groups selected by lane>>4).
template<int NT, int MT>
__global__ __launch_bounds__(128)
void k_gemm_wmma(const float* __restrict__ A, const _Float16* __restrict__ pb,
                 float* __restrict__ H, int K) {
    const int C = NT * 16;
    int wave = blockIdx.x * (blockDim.x >> 5) + (threadIdx.x >> 5);
    int lane = threadIdx.x & 31;
    int row0 = wave * 16 * MT;
    if (row0 >= N_NODES) return;          // wave-uniform: EXEC stays all-ones
    int half = lane >> 4;

    const float* arow[MT];
#pragma unroll
    for (int mt = 0; mt < MT; ++mt)
        arow[mt] = A + (size_t)(row0 + mt * 16 + (lane & 15)) * K;

    v8f zero = {0.f,0.f,0.f,0.f,0.f,0.f,0.f,0.f};
    v8f acc[MT][NT];
#pragma unroll
    for (int mt = 0; mt < MT; ++mt)
#pragma unroll
        for (int nt = 0; nt < NT; ++nt) acc[mt][nt] = zero;

    int full = K >> 5;
    for (int kc = 0; kc < full; ++kc) {
        int kb = (kc << 5) + half * 8;
        v16h b[NT];
#pragma unroll
        for (int nt = 0; nt < NT; ++nt)
            b[nt] = *(const v16h*)(pb + ((size_t)(kc * NT + nt) * 32 + lane) * 16);
#pragma unroll
        for (int mt = 0; mt < MT; ++mt) {
            // prefetch next K-chunk of this row group (global_prefetch_b8)
            if (kc + 1 < full) __builtin_prefetch(arow[mt] + kb + 32, 0, 1);
            v16h a;
#pragma unroll
            for (int j = 0; j < 8; ++j) a[j]     = (_Float16)arow[mt][kb + j];
#pragma unroll
            for (int j = 0; j < 8; ++j) a[j + 8] = (_Float16)arow[mt][kb + 16 + j];
#pragma unroll
            for (int nt = 0; nt < NT; ++nt)
                acc[mt][nt] = __builtin_amdgcn_wmma_f32_16x16x32_f16(
                    false, a, false, b[nt], (short)0, acc[mt][nt], false, false);
        }
    }
    if (K & 31) {                          // zero-masked tail chunk (K=932)
        int kc = full;
        v16h b[NT];
#pragma unroll
        for (int nt = 0; nt < NT; ++nt)
            b[nt] = *(const v16h*)(pb + ((size_t)(kc * NT + nt) * 32 + lane) * 16);
#pragma unroll
        for (int mt = 0; mt < MT; ++mt) {
            v16h a;
#pragma unroll
            for (int j = 0; j < 16; ++j) {
                int k = (kc << 5) + half * 8 + j + ((j >= 8) ? 8 : 0);
                a[j] = (_Float16)((k < K) ? arow[mt][k] : 0.0f);
            }
#pragma unroll
            for (int nt = 0; nt < NT; ++nt)
                acc[mt][nt] = __builtin_amdgcn_wmma_f32_16x16x32_f16(
                    false, a, false, b[nt], (short)0, acc[mt][nt], false, false);
        }
    }
    // C/D layout: lane holds N = lane&15, VGPR v holds row v + 8*(lane>>4)
#pragma unroll
    for (int mt = 0; mt < MT; ++mt)
#pragma unroll
        for (int nt = 0; nt < NT; ++nt) {
            int n = nt * 16 + (lane & 15);
#pragma unroll
            for (int v = 0; v < 8; ++v) {
                int m = row0 + mt * 16 + v + half * 8;
                H[(size_t)m * C + n] = acc[mt][nt][v];
            }
        }
}

// ---------------- aggregation ----------------
// Self-loop term initializes the output: out[i,c] = H[i,c] * dinv[i]^2
__global__ void k_agg_self(const float* __restrict__ H,
                           const float* __restrict__ dinv,
                           float* __restrict__ out, int C) {
    int idx = blockIdx.x * blockDim.x + threadIdx.x;
    if (idx >= N_NODES * C) return;
    int i = idx / C;
    float di = dinv[i];
    out[idx] = H[idx] * di * di;
}

// Edge scatter-add: one thread per (edge, 4-channel group); L2-resident atomics
template<int C>
__global__ void k_agg_edges(const int* __restrict__ src, const int* __restrict__ dst,
                            const float* __restrict__ dinv,
                            const float* __restrict__ H, float* out) {
    constexpr int G = C / 4;
    int t = blockIdx.x * blockDim.x + threadIdx.x;
    if (t >= N_EDGES * G) return;
    int e = t / G;
    int q = t - e * G;
    int s = src[e], d = dst[e];
    float nr = dinv[s] * dinv[d];
    const float4 v = *(const float4*)(H + (size_t)s * C + q * 4);
    float* op = out + (size_t)d * C + q * 4;
    atomicAdd(op + 0, v.x * nr);
    atomicAdd(op + 1, v.y * nr);
    atomicAdd(op + 2, v.z * nr);
    atomicAdd(op + 3, v.w * nr);
}

__global__ void k_bias_act(float* X, const float* __restrict__ b, int C, int elu) {
    int idx = blockIdx.x * blockDim.x + threadIdx.x;
    if (idx >= N_NODES * C) return;
    int c = idx % C;
    float v = X[idx] + b[c];
    if (elu) v = (v > 0.0f) ? v : expm1f(v);   // jax.nn.elu, alpha=1
    X[idx] = v;
}

// ---------------- launch ----------------
static inline size_t align256(size_t v) { return (v + 255) & ~(size_t)255; }

extern "C" void kernel_launch(void* const* d_in, const int* in_sizes, int n_in,
                              void* d_out, int out_size, void* d_ws, size_t ws_size,
                              hipStream_t stream) {
    const float* x  = (const float*)d_in[0];
    const int*   ei = (const int*)  d_in[1];
    const float* W1 = (const float*)d_in[2];
    const float* b1 = (const float*)d_in[3];
    const float* W2 = (const float*)d_in[4];
    const float* b2 = (const float*)d_in[5];
    const float* W3 = (const float*)d_in[6];
    const float* b3 = (const float*)d_in[7];
    const int* src = ei;
    const int* dst = ei + N_EDGES;
    float* out = (float*)d_out;

    char* ws = (char*)d_ws;
    size_t off = 0;
    float* dinv = (float*)(ws + off); off = align256(off + (size_t)N_NODES * 4);
    float* bufH = (float*)(ws + off); off = align256(off + (size_t)N_NODES * C_HID * 4);
    float* bufA = (float*)(ws + off); off = align256(off + (size_t)N_NODES * C_HID * 4);
    const int CH1 = (C_IN + 31) / 32;                      // 30 K-chunks
    _Float16* pb1 = (_Float16*)(ws + off); off = align256(off + (size_t)CH1 * 2 * 512 * 2);
    _Float16* pb2 = (_Float16*)(ws + off); off = align256(off + (size_t)1   * 2 * 512 * 2);
    _Float16* pb3 = (_Float16*)(ws + off); off = align256(off + (size_t)1   * 1 * 512 * 2);

    const int TPB = 256;
    // --- normalization: deg -> dinv (in place) ---
    k_deg_init <<<(N_NODES + TPB - 1) / TPB, TPB, 0, stream>>>(dinv);
    k_deg_edges<<<(N_EDGES + TPB - 1) / TPB, TPB, 0, stream>>>(dst, dinv);
    k_dinv     <<<(N_NODES + TPB - 1) / TPB, TPB, 0, stream>>>(dinv);

    // --- pack weights into f16 WMMA B fragments ---
    k_pack_B<<<(CH1 * 2 * 512 + TPB - 1) / TPB, TPB, 0, stream>>>(W1, pb1, C_IN,  C_HID, CH1);
    k_pack_B<<<(2 * 512 + TPB - 1) / TPB,       TPB, 0, stream>>>(W2, pb2, C_HID, C_HID, 1);
    k_pack_B<<<(1 * 512 + TPB - 1) / TPB,       TPB, 0, stream>>>(W3, pb3, C_HID, C_OUT, 1);

    const int MT    = 5;                     // 100000 = 16 * 5 * 1250 exactly
    const int WAVES = N_NODES / (16 * MT);   // 1250
    const int GB    = (WAVES + 3) / 4;       // 4 waves / 128-thread block
    const int NHID  = N_NODES * C_HID;
    const int NOUT  = N_NODES * C_OUT;
    const int EHID  = N_EDGES * (C_HID / 4);
    const int EOUT  = N_EDGES * (C_OUT / 4);

    // --- layer 1: X(100000x932) @ W1 -> agg -> +b1, ELU ---
    k_gemm_wmma<2,5><<<GB, 128, 0, stream>>>(x, pb1, bufH, C_IN);
    k_agg_self <<<(NHID + TPB - 1) / TPB, TPB, 0, stream>>>(bufH, dinv, bufA, C_HID);
    k_agg_edges<C_HID><<<(EHID + TPB - 1) / TPB, TPB, 0, stream>>>(src, dst, dinv, bufH, bufA);
    k_bias_act <<<(NHID + TPB - 1) / TPB, TPB, 0, stream>>>(bufA, b1, C_HID, 1);

    // --- layer 2 ---
    k_gemm_wmma<2,5><<<GB, 128, 0, stream>>>(bufA, pb2, bufH, C_HID);
    k_agg_self <<<(NHID + TPB - 1) / TPB, TPB, 0, stream>>>(bufH, dinv, bufA, C_HID);
    k_agg_edges<C_HID><<<(EHID + TPB - 1) / TPB, TPB, 0, stream>>>(src, dst, dinv, bufH, bufA);
    k_bias_act <<<(NHID + TPB - 1) / TPB, TPB, 0, stream>>>(bufA, b2, C_HID, 1);

    // --- layer 3 (no ELU), writes final output ---
    k_gemm_wmma<1,5><<<GB, 128, 0, stream>>>(bufA, pb3, bufH, C_HID);
    k_agg_self <<<(NOUT + TPB - 1) / TPB, TPB, 0, stream>>>(bufH, dinv, out, C_OUT);
    k_agg_edges<C_OUT><<<(EOUT + TPB - 1) / TPB, TPB, 0, stream>>>(src, dst, dinv, bufH, out);
    k_bias_act <<<(NOUT + TPB - 1) / TPB, TPB, 0, stream>>>(out, b3, C_OUT, 0);
}
